// Attention_39651138076722
// MI455X (gfx1250) — matrix operand order repaired and checked by reading, not compiled, not run on hardware
//
#include <hip/hip_runtime.h>
#include <hip/hip_bf16.h>
#include <stdint.h>
#include <math.h>

// MI455X / gfx1250: wave32, WMMA f16 16x16x32 (fp32 accum) path.
typedef __attribute__((ext_vector_type(16))) _Float16 v16h;
typedef __attribute__((ext_vector_type(8)))  float    v8f;

#define B_    4
#define C_    128
#define NH    4
#define DH    32
#define NPIX  4096
// log2(e) / sqrt(dh): scores land in log2 domain -> softmax uses raw v_exp_f32
#define SCALE_LOG2 0.25504168f

#define WMMA16(a, b, c) __builtin_amdgcn_wmma_f32_16x16x32_f16(false, (a), false, (b), (short)0, (c), false, false)

// Load 8 halves from p0 into low half, 8 halves from p1 into high half (A-operand layout).
static __device__ __forceinline__ v16h ld16h(const _Float16* p0, const _Float16* p1) {
  union { v16h h; uint4 u[2]; } u;
  u.u[0] = *(const uint4*)p0;
  u.u[1] = *(const uint4*)p1;
  return u.h;
}
// Load 8 contiguous dwords (16 packed halves) -> B-operand layout.
static __device__ __forceinline__ v16h ld16h_b(const uint32_t* p) {
  union { v16h h; uint4 u[2]; } u;
  const uint4* q = (const uint4*)p;
  u.u[0] = q[0];
  u.u[1] = q[1];
  return u.h;
}

// --------------------------------------------------------------------------
// Kernel 1: per-pixel channel norm (mean/var over C=128), write h^T (B,N,C) f16
// --------------------------------------------------------------------------
__global__ void norm_kernel(const float* __restrict__ x,
                            const float* __restrict__ nw,
                            const float* __restrict__ nb,
                            _Float16* __restrict__ hN) {
  int pix = blockIdx.x * blockDim.x + threadIdx.x;   // 0..16383 (B*N)
  int b = pix >> 12;
  int n = pix & 4095;
  const float* xp = x + (size_t)b * C_ * NPIX + n;
  float s = 0.f, s2 = 0.f;
  #pragma unroll 8
  for (int c = 0; c < C_; ++c) {
    float v = xp[(size_t)c * NPIX];
    s += v; s2 += v * v;
  }
  float mean = s * (1.0f / C_);
  float var  = s2 * (1.0f / C_) - mean * mean;
  float rstd = rsqrtf(var + 1e-6f);
  _Float16* hp = hN + (size_t)pix * C_;
  #pragma unroll 8
  for (int c = 0; c < C_; ++c) {
    float v = (xp[(size_t)c * NPIX] - mean) * rstd * nw[c] + nb[c];
    hp[c] = (_Float16)v;
  }
}

// --------------------------------------------------------------------------
// Kernel 2: QKV GEMM  (3C,C)x(C,N) per batch via WMMA f16; scatter into
//   qT (B,NH,N,DH)*scale_log2, kT (B,NH,N,DH), v (B,NH,DH,N)   all f16
// --------------------------------------------------------------------------
__global__ __launch_bounds__(128) void qkv_kernel(const _Float16* __restrict__ hN,
                           const float* __restrict__ qkv_w,
                           const float* __restrict__ qkv_b,
                           _Float16* __restrict__ qT,
                           _Float16* __restrict__ kT,
                           _Float16* __restrict__ vM) {
  int wid  = (blockIdx.x * blockDim.x + threadIdx.x) >> 5;
  int lane = threadIdx.x & 31;
  int ln = lane & 15, hi = lane >> 4;
  int b  = wid / (24 * 256);
  int r  = wid - b * 24 * 256;
  int to = r >> 8;                 // 0..23 (384/16 output-row tiles)
  int tn = r & 255;                // 0..255 (4096/16 column tiles)
  int o  = to * 16 + ln;
  int n  = tn * 16 + ln;

  const uint32_t* hrow = (const uint32_t*)(hN + ((size_t)b * NPIX + n) * C_);
  v8f acc = {};
  #pragma unroll
  for (int kc = 0; kc < 4; ++kc) {
    int c0 = kc * 32 + hi * 8;
    const float* w0 = qkv_w + (size_t)o * C_ + c0;
    v16h a;
    #pragma unroll
    for (int i = 0; i < 8; ++i) { a[i] = (_Float16)w0[i]; a[8 + i] = (_Float16)w0[16 + i]; }
    v16h bm = ld16h((const _Float16*)(hrow + kc * 16 + hi * 4), (const _Float16*)(hrow + kc * 16 + hi * 4 + 8));
    acc = WMMA16(a, bm, acc);
  }
  int which = to >> 3;             // 0=q 1=k 2=v (uniform per wave)
  int head  = (to & 7) >> 1;
  int cbase = (to & 1) * 16;
  size_t bh = (size_t)(b * NH + head);
  #pragma unroll
  for (int rr = 0; rr < 8; ++rr) {
    int rowin = rr + 8 * hi;                 // row within 16-tile
    float val = acc[rr] + qkv_b[to * 16 + rowin];
    int c = cbase + rowin;                   // channel within head (0..31)
    if (which == 0)      qT[(bh * NPIX + n) * DH + c] = (_Float16)(val * SCALE_LOG2);
    else if (which == 1) kT[(bh * NPIX + n) * DH + c] = (_Float16)val;
    else                 vM[(bh * DH + c) * NPIX + n] = (_Float16)val;
  }
}

// --------------------------------------------------------------------------
// Kernel 3: fused flash attention.
//  - block = 4 waves = 128 query rows of one (b,head); wave owns 2 q-tiles.
//  - keys streamed in chunks of 32; S^T = K^T x Q so softmax stats are per-lane.
//  - V chunk staged block-cooperatively into LDS via global_load_async_to_lds
//    (ASYNCcnt, double-buffered), overlapped with gemm1 + softmax.
//  - P re-layout (C-layout f32 -> B-operand f16) through padded LDS rows.
// --------------------------------------------------------------------------
#define VSTRIDE 40   // halves per LDS row (80B): conflict-free b128 reads
__global__ __launch_bounds__(128) void attn_kernel(const _Float16* __restrict__ qT,
                            const _Float16* __restrict__ kT,
                            const _Float16* __restrict__ vM,
                            _Float16* __restrict__ hA) {
  __shared__ __align__(16) _Float16 VB[2][32 * VSTRIDE];      // V double buffer
  __shared__ __align__(16) _Float16 PB[4][2][16 * VSTRIDE];   // per-wave, per-qtile P
  int w    = threadIdx.x >> 5;
  int lane = threadIdx.x & 31;
  int ln = lane & 15, hi = lane >> 4;
  int bh    = blockIdx.x >> 5;                 // 0..15  (B*NH)
  int nbase = (blockIdx.x & 31) * 128;         // block's 128 query rows
  int n0    = nbase + w * 32;                  // wave's 32 rows (2 tiles)

  // Q tiles as B-operands (32 x 16), loaded once.
  v16h bq0 = ld16h(qT + ((size_t)bh * NPIX + n0 + ln) * DH + hi * 8, qT + ((size_t)bh * NPIX + n0 + ln) * DH + hi * 8 + 16);
  v16h bq1 = ld16h(qT + ((size_t)bh * NPIX + n0 + 16 + ln) * DH + hi * 8, qT + ((size_t)bh * NPIX + n0 + 16 + ln) * DH + hi * 8 + 16);

  float m0r = -1e30f, l0r = 0.f, m1r = -1e30f, l1r = 0.f;
  v8f O00 = {}, O01 = {}, O10 = {}, O11 = {};

  // cooperative async V staging: thread f covers row c=f>>2, cols m0=(f&3)*8
  int f  = threadIdx.x;
  int vc = f >> 2, vm = (f & 3) * 8;
  const _Float16* vsrc = vM + ((size_t)bh * DH + vc) * NPIX + vm;
  uint32_t vb0 = (uint32_t)(uintptr_t)&VB[0][0] + (uint32_t)(vc * VSTRIDE + vm) * 2;
  uint32_t vb1 = (uint32_t)(uintptr_t)&VB[1][0] + (uint32_t)(vc * VSTRIDE + vm) * 2;

  // prologue: stage chunk 0 into buffer 0 (ASYNCcnt)
  {
    uint64_t g = (uint64_t)(uintptr_t)vsrc;
    asm volatile("global_load_async_to_lds_b128 %0, %1, off" :: "v"(vb0), "v"(g) : "memory");
  }

  _Float16* P0 = &PB[w][0][0];
  _Float16* P1 = &PB[w][1][0];

  for (int it = 0; it < NPIX / 32; ++it) {
    int mc = it * 32;
    // current V chunk complete (wave-local), then block-wide publish
    asm volatile("s_wait_asynccnt 0" ::: "memory");
    asm volatile("s_wait_dscnt 0" ::: "memory");
    __syncthreads();
    // kick off next chunk into the other buffer; hides under gemm1+softmax
    if (it + 1 < NPIX / 32) {
      uint64_t g = (uint64_t)(uintptr_t)(vsrc + mc + 32);
      uint32_t d = (it & 1) ? vb0 : vb1;
      asm volatile("global_load_async_to_lds_b128 %0, %1, off" :: "v"(d), "v"(g) : "memory");
    }

    // --- gemm1: S^T tiles (keys on rows), shared across both q-tiles ---
    const _Float16* k0p = kT + ((size_t)bh * NPIX + mc + ln) * DH + hi * 8;
    v16h a0 = ld16h(k0p, k0p + 16);
    const _Float16* k1p = k0p + 16 * DH;
    v16h a1 = ld16h(k1p, k1p + 16);
    v8f S00 = {}, S01 = {}, S10 = {}, S11 = {};
    S00 = WMMA16(a0, bq0, S00);
    S01 = WMMA16(a1, bq0, S01);
    S10 = WMMA16(a0, bq1, S10);
    S11 = WMMA16(a1, bq1, S11);

    // --- online softmax, q-tile 0 (scores already in log2 domain) ---
    {
      float cm = -1e30f;
      #pragma unroll
      for (int i = 0; i < 8; ++i) { cm = fmaxf(cm, S00[i]); cm = fmaxf(cm, S01[i]); }
      cm = fmaxf(cm, __shfl_xor(cm, 16, 32));
      float mn = fmaxf(m0r, cm);
      float corr = exp2f(m0r - mn);
      float ps = 0.f;
      #pragma unroll
      for (int i = 0; i < 8; i += 2) {
        float pa = exp2f(S00[i] - mn),     pb = exp2f(S00[i + 1] - mn);
        float pc = exp2f(S01[i] - mn),     pd = exp2f(S01[i + 1] - mn);
        ps += pa + pb + pc + pd;
        union { _Float16 h[2]; uint32_t u; } k1v, k2v;
        k1v.h[0] = (_Float16)pa; k1v.h[1] = (_Float16)pb;
        k2v.h[0] = (_Float16)pc; k2v.h[1] = (_Float16)pd;
        *(uint32_t*)&P0[ln * VSTRIDE + 8 * hi + i]      = k1v.u;
        *(uint32_t*)&P0[ln * VSTRIDE + 16 + 8 * hi + i] = k2v.u;
      }
      ps += __shfl_xor(ps, 16, 32);
      l0r = l0r * corr + ps;
      m0r = mn;
      #pragma unroll
      for (int i = 0; i < 8; ++i) { O00[i] *= corr; O01[i] *= corr; }
    }
    // --- online softmax, q-tile 1 ---
    {
      float cm = -1e30f;
      #pragma unroll
      for (int i = 0; i < 8; ++i) { cm = fmaxf(cm, S10[i]); cm = fmaxf(cm, S11[i]); }
      cm = fmaxf(cm, __shfl_xor(cm, 16, 32));
      float mn = fmaxf(m1r, cm);
      float corr = exp2f(m1r - mn);
      float ps = 0.f;
      #pragma unroll
      for (int i = 0; i < 8; i += 2) {
        float pa = exp2f(S10[i] - mn),     pb = exp2f(S10[i + 1] - mn);
        float pc = exp2f(S11[i] - mn),     pd = exp2f(S11[i + 1] - mn);
        ps += pa + pb + pc + pd;
        union { _Float16 h[2]; uint32_t u; } k1v, k2v;
        k1v.h[0] = (_Float16)pa; k1v.h[1] = (_Float16)pb;
        k2v.h[0] = (_Float16)pc; k2v.h[1] = (_Float16)pd;
        *(uint32_t*)&P1[ln * VSTRIDE + 8 * hi + i]      = k1v.u;
        *(uint32_t*)&P1[ln * VSTRIDE + 16 + 8 * hi + i] = k2v.u;
      }
      ps += __shfl_xor(ps, 16, 32);
      l1r = l1r * corr + ps;
      m1r = mn;
      #pragma unroll
      for (int i = 0; i < 8; ++i) { O10[i] *= corr; O11[i] *= corr; }
    }

    // same-wave DS store->load ordering for P
    asm volatile("s_wait_dscnt 0" ::: "memory");

    // --- gemm2: O += V x P^T  (V from LDS, staged async) ---
    v16h bp0 = ld16h(P0 + ln * VSTRIDE + hi * 8, P0 + ln * VSTRIDE + hi * 8 + 16);
    v16h bp1 = ld16h(P1 + ln * VSTRIDE + hi * 8, P1 + ln * VSTRIDE + hi * 8 + 16);
    const _Float16* Vb = &VB[it & 1][0];
    const _Float16* vv0 = Vb + ln * VSTRIDE + hi * 8;
    v16h av0 = ld16h(vv0, vv0 + 16);
    const _Float16* vv1 = Vb + (ln + 16) * VSTRIDE + hi * 8;
    v16h av1 = ld16h(vv1, vv1 + 16);
    O00 = WMMA16(av0, bp0, O00);
    O01 = WMMA16(av1, bp0, O01);
    O10 = WMMA16(av0, bp1, O10);
    O11 = WMMA16(av1, bp1, O11);
  }

  int head = bh & 3, b = bh >> 2;
  float inv0 = 1.0f / l0r;
  _Float16* hrow0 = hA + ((size_t)b * NPIX + n0 + ln) * C_ + head * DH;
  float inv1 = 1.0f / l1r;
  _Float16* hrow1 = hA + ((size_t)b * NPIX + n0 + 16 + ln) * C_ + head * DH;
  #pragma unroll
  for (int i = 0; i < 8; ++i) {
    hrow0[i + 8 * hi]      = (_Float16)(O00[i] * inv0);
    hrow0[16 + i + 8 * hi] = (_Float16)(O01[i] * inv0);
    hrow1[i + 8 * hi]      = (_Float16)(O10[i] * inv1);
    hrow1[16 + i + 8 * hi] = (_Float16)(O11[i] * inv1);
  }
}

// --------------------------------------------------------------------------
// Kernel 4: output projection (C,C)x(C,N) + bias + residual, fp32 out.
// --------------------------------------------------------------------------
__global__ __launch_bounds__(128) void proj_kernel(const _Float16* __restrict__ hA,
                            const float* __restrict__ proj_w,
                            const float* __restrict__ proj_b,
                            const float* __restrict__ x,
                            float* __restrict__ out) {
  int wid  = (blockIdx.x * blockDim.x + threadIdx.x) >> 5;
  int lane = threadIdx.x & 31;
  int ln = lane & 15, hi = lane >> 4;
  int b  = wid / (8 * 256);
  int r  = wid - b * 8 * 256;
  int to = r >> 8;                 // 0..7
  int tn = r & 255;
  int o  = to * 16 + ln;
  int n  = tn * 16 + ln;

  const uint32_t* hrow = (const uint32_t*)(hA + ((size_t)b * NPIX + n) * C_);
  v8f acc = {};
  #pragma unroll
  for (int kc = 0; kc < 4; ++kc) {
    int c0 = kc * 32 + hi * 8;
    const float* w0 = proj_w + (size_t)o * C_ + c0;
    v16h a;
    #pragma unroll
    for (int i = 0; i < 8; ++i) { a[i] = (_Float16)w0[i]; a[8 + i] = (_Float16)w0[16 + i]; }
    v16h bm = ld16h((const _Float16*)(hrow + kc * 16 + hi * 4), (const _Float16*)(hrow + kc * 16 + hi * 4 + 8));
    acc = WMMA16(a, bm, acc);
  }
  #pragma unroll
  for (int rr = 0; rr < 8; ++rr) {
    int orow = to * 16 + rr + 8 * hi;
    size_t idx = ((size_t)b * C_ + orow) * NPIX + n;
    out[idx] = x[idx] + acc[rr] + proj_b[orow];
  }
}

// --------------------------------------------------------------------------
extern "C" void kernel_launch(void* const* d_in, const int* in_sizes, int n_in,
                              void* d_out, int out_size, void* d_ws, size_t ws_size,
                              hipStream_t stream) {
  const float* x      = (const float*)d_in[0];
  const float* norm_w = (const float*)d_in[1];
  const float* norm_b = (const float*)d_in[2];
  const float* qkv_w  = (const float*)d_in[3];
  const float* qkv_b  = (const float*)d_in[4];
  const float* proj_w = (const float*)d_in[5];
  const float* proj_b = (const float*)d_in[6];

  char* ws = (char*)d_ws;
  const size_t SEG = (size_t)B_ * NPIX * C_ * sizeof(_Float16); // 4 MiB each
  _Float16* hN = (_Float16*)(ws);            // normed input, (B,N,C)
  _Float16* qT = (_Float16*)(ws + 1 * SEG);  // (B,NH,N,DH), pre-scaled by log2e/sqrt(dh)
  _Float16* kT = (_Float16*)(ws + 2 * SEG);  // (B,NH,N,DH)
  _Float16* vM = (_Float16*)(ws + 3 * SEG);  // (B,NH,DH,N)
  _Float16* hA = (_Float16*)(ws + 4 * SEG);  // attention out, (B,N,C)

  norm_kernel<<<64,   256, 0, stream>>>(x, norm_w, norm_b, hN);
  qkv_kernel <<<6144, 128, 0, stream>>>(hN, qkv_w, qkv_b, qT, kT, vM);
  attn_kernel<<<512,  128, 0, stream>>>(qT, kT, vM, hA);
  proj_kernel<<<2048, 128, 0, stream>>>(hA, proj_w, proj_b, x, (float*)d_out);
}
